// GCNMLP_76192719832099
// MI455X (gfx1250) — matrix operand order repaired and checked by reading, not compiled, not run on hardware
//
#include <hip/hip_runtime.h>
#include <hip/hip_bf16.h>

#define NB    8
#define VN    4608
#define NV    36864          // NB*VN
#define EPER  73728
#define EE    589824         // NB*EPER
#define EP    626688         // EE + NV (with self loops)
#define CC    512
#define NCLS  3
#define LN_EPS 1e-5f

typedef __attribute__((ext_vector_type(16))) __bf16 v16bf;
typedef __attribute__((ext_vector_type(8)))  float  v8f;

union Frag { uint4 u[2]; v16bf v; };

__device__ __forceinline__ unsigned short f2bf(float f) {
    unsigned int u = __float_as_uint(f);
    u += 0x7FFFu + ((u >> 16) & 1u);      // round-to-nearest-even
    return (unsigned short)(u >> 16);
}
__device__ __forceinline__ float bf2f(unsigned short h) {
    return __uint_as_float(((unsigned int)h) << 16);
}

template <int BS>
__device__ __forceinline__ float block_reduce(float v, float* s) {
    int tid = threadIdx.x;
    s[tid] = v; __syncthreads();
    #pragma unroll
    for (int o = BS >> 1; o > 0; o >>= 1) {
        if (tid < o) s[tid] += s[tid + o];
        __syncthreads();
    }
    float r = s[0]; __syncthreads();
    return r;
}

// ---------------- CSR build ----------------
__global__ void zero_kernel(float* deg, int* cnt, int* cnt2) {
    int i = blockIdx.x * blockDim.x + threadIdx.x;
    if (i < NV) { deg[i] = 0.f; cnt[i] = 0; cnt2[i] = 0; }
}

__global__ void edge_setup_kernel(const int* __restrict__ ei, const float* __restrict__ ew,
                                  int* srcf, int* tgtf, float* ewf, float* deg, int* cnt) {
    int e = blockIdx.x * blockDim.x + threadIdx.x;
    if (e >= EP) return;
    int s, t; float w;
    if (e < EE) {
        int off = (e / EPER) * VN;
        s = ei[e] + off;
        t = ei[EE + e] + off;
        w = ew[e];
    } else {
        s = t = e - EE; w = 1.0f;
    }
    srcf[e] = s; tgtf[e] = t; ewf[e] = w;
    atomicAdd(&deg[t], w);
    atomicAdd(&cnt[t], 1);
}

__global__ __launch_bounds__(1024) void scan_kernel(const int* __restrict__ cnt, int* __restrict__ row_ptr) {
    __shared__ int sdata[1024];
    __shared__ int carry;
    int tid = threadIdx.x;
    if (tid == 0) carry = 0;
    __syncthreads();
    for (int base = 0; base < NV; base += 1024) {
        int v = cnt[base + tid];
        sdata[tid] = v; __syncthreads();
        #pragma unroll
        for (int o = 1; o < 1024; o <<= 1) {
            int t = (tid >= o) ? sdata[tid - o] : 0;
            __syncthreads();
            sdata[tid] += t;
            __syncthreads();
        }
        row_ptr[base + tid] = carry + sdata[tid] - v;   // exclusive
        __syncthreads();
        if (tid == 1023) carry += sdata[1023];
        __syncthreads();
    }
    if (tid == 0) row_ptr[NV] = carry;
}

__global__ void dinv_kernel(const float* __restrict__ deg, float* __restrict__ dinv) {
    int i = blockIdx.x * blockDim.x + threadIdx.x;
    if (i < NV) dinv[i] = deg[i] > 0.f ? rsqrtf(deg[i]) : 0.f;
}

__global__ void csr_fill_kernel(const int* __restrict__ srcf, const int* __restrict__ tgtf,
                                const float* __restrict__ ewf, const float* __restrict__ dinv,
                                const int* __restrict__ row_ptr, int* cnt2,
                                int* __restrict__ csr_src, float* __restrict__ csr_w) {
    int e = blockIdx.x * blockDim.x + threadIdx.x;
    if (e >= EP) return;
    int t = tgtf[e], s = srcf[e];
    int pos = row_ptr[t] + atomicAdd(&cnt2[t], 1);
    csr_src[pos] = s;
    csr_w[pos]   = dinv[s] * ewf[e] * dinv[t];
}

// ---------------- weight pack into WMMA B-fragment layout ----------------
// Fragment (kt, nf): N=nf*16+(lane&15), h=lane>>4 ; el<8 -> K=kt*32+8h+el ; el>=8 -> K=kt*32+16+8h+(el-8)
__global__ void packb_kernel(const float* __restrict__ W, unsigned short* __restrict__ Bp) {
    int tid = blockIdx.x * blockDim.x + threadIdx.x;
    if (tid >= CC * CC) return;
    int el   = tid & 15;
    int lane = (tid >> 4) & 31;
    int frag = tid >> 9;
    int nf   = frag % (CC / 16);
    int kt   = frag / (CC / 16);
    int k = kt * 32 + ((lane >> 4) << 3) + (el & 7) + ((el >> 3) << 4);
    int n = nf * 16 + (lane & 15);
    Bp[tid] = f2bf(W[k * CC + n]);
}

// ---------------- layer 0 (D_IN = 1, rank-1) : agg scalar + bias/relu/LN ----------------
__global__ __launch_bounds__(256) void layer0_kernel(const float* __restrict__ x,
                                                     const float* __restrict__ w0, const float* __restrict__ b0,
                                                     const float* __restrict__ ln_g, const float* __restrict__ ln_b,
                                                     const int* __restrict__ row_ptr, const int* __restrict__ csr_src,
                                                     const float* __restrict__ csr_w,
                                                     unsigned short* __restrict__ hbf) {
    __shared__ float sred[256];
    int i = blockIdx.x, tid = threadIdx.x;
    int beg = row_ptr[i], end = row_ptr[i + 1];
    float p = 0.f;
    for (int e = beg + tid; e < end; e += 256) p += csr_w[e] * x[csr_src[e]];
    float tval = block_reduce<256>(p, sred);
    int c0 = tid * 2;
    float v0 = fmaxf(tval * w0[c0]     + b0[c0],     0.f);
    float v1 = fmaxf(tval * w0[c0 + 1] + b0[c0 + 1], 0.f);
    float sum = block_reduce<256>(v0 + v1, sred);
    float sq  = block_reduce<256>(v0 * v0 + v1 * v1, sred);
    float mu = sum * (1.f / CC);
    float var = sq * (1.f / CC) - mu * mu;
    float rs = rsqrtf(var + LN_EPS);
    hbf[(size_t)i * CC + c0]     = f2bf((v0 - mu) * rs * ln_g[c0]     + ln_b[c0]);
    hbf[(size_t)i * CC + c0 + 1] = f2bf((v1 - mu) * rs * ln_g[c0 + 1] + ln_b[c0 + 1]);
}

// ---------------- WMMA GEMM: hw[NV,512] = hbf[NV,512](bf16) @ Bp(packed bf16), fp32 out ----------------
__global__ __launch_bounds__(256) void gemm_bf16_kernel(const unsigned short* __restrict__ A,
                                                        const unsigned short* __restrict__ Bp,
                                                        float* __restrict__ Out) {
    const int lane = threadIdx.x & 31;
    const int wave = blockIdx.x * (blockDim.x >> 5) + (threadIdx.x >> 5);
    const int NT = CC / 32;             // 16 n-tiles of width 32
    const int mt = wave / NT;
    const int nt = wave % NT;
    const int m0 = mt * 64;
    const int n0 = nt * 32;
    const int halfsel = lane >> 4;      // 0 or 1
    const int rowInTile = lane & 15;

    v8f acc[4][2];
    #pragma unroll
    for (int r = 0; r < 4; ++r)
        #pragma unroll
        for (int c = 0; c < 2; ++c)
            #pragma unroll
            for (int j = 0; j < 8; ++j) acc[r][c][j] = 0.f;

    for (int kk = 0; kk < CC; kk += 32) {
        Frag a[4];
        const int colbase = kk + (halfsel << 3);
        #pragma unroll
        for (int r = 0; r < 4; ++r) {
            const unsigned short* p = A + (size_t)(m0 + r * 16 + rowInTile) * CC + colbase;
            a[r].u[0] = *(const uint4*)(p);
            a[r].u[1] = *(const uint4*)(p + 16);
        }
        Frag b[2];
        #pragma unroll
        for (int c = 0; c < 2; ++c) {
            const int frag = (kk >> 5) * (CC / 16) + (n0 / 16 + c);
            const uint4* q = (const uint4*)(Bp + ((size_t)frag * 32 + lane) * 16);
            b[c].u[0] = q[0];
            b[c].u[1] = q[1];
        }
        #pragma unroll
        for (int r = 0; r < 4; ++r)
            #pragma unroll
            for (int c = 0; c < 2; ++c)
                acc[r][c] = __builtin_amdgcn_wmma_f32_16x16x32_bf16(
                    false, a[r].v, false, b[c].v, (short)0, acc[r][c], false, false);
    }
    // C/D layout: lanes 0-15 -> N=lane, M=j ; lanes 16-31 -> N=lane-16, M=8+j
    #pragma unroll
    for (int r = 0; r < 4; ++r) {
        const int rb = m0 + r * 16 + (halfsel << 3);
        #pragma unroll
        for (int c = 0; c < 2; ++c) {
            const int col = n0 + c * 16 + rowInTile;
            #pragma unroll
            for (int j = 0; j < 8; ++j)
                Out[(size_t)(rb + j) * CC + col] = acc[r][c][j];
        }
    }
}

// ---------------- CSR aggregation + bias + relu + LN -> bf16 ----------------
__global__ __launch_bounds__(256) void agg_ln_kernel(const float* __restrict__ hw,
                                                     const float* __restrict__ bias,
                                                     const float* __restrict__ ln_g, const float* __restrict__ ln_b,
                                                     const int* __restrict__ row_ptr, const int* __restrict__ csr_src,
                                                     const float* __restrict__ csr_w,
                                                     unsigned short* __restrict__ hbf) {
    __shared__ float sred[256];
    int i = blockIdx.x, tid = threadIdx.x;
    int beg = row_ptr[i], end = row_ptr[i + 1];
    int c0 = tid * 2;
    float a0 = 0.f, a1 = 0.f;
    for (int e = beg; e < end; ++e) {
        float w = csr_w[e];
        int   s = csr_src[e];
        const float2 hv = *(const float2*)(hw + (size_t)s * CC + c0);
        a0 += w * hv.x;
        a1 += w * hv.y;
    }
    a0 = fmaxf(a0 + bias[c0],     0.f);
    a1 = fmaxf(a1 + bias[c0 + 1], 0.f);
    float sum = block_reduce<256>(a0 + a1, sred);
    float sq  = block_reduce<256>(a0 * a0 + a1 * a1, sred);
    float mu = sum * (1.f / CC);
    float var = sq * (1.f / CC) - mu * mu;
    float rs = rsqrtf(var + LN_EPS);
    hbf[(size_t)i * CC + c0]     = f2bf((a0 - mu) * rs * ln_g[c0]     + ln_b[c0]);
    hbf[(size_t)i * CC + c0 + 1] = f2bf((a1 - mu) * rs * ln_g[c0 + 1] + ln_b[c0 + 1]);
}

// ---------------- max-pool over nodes + LN + head ----------------
__global__ __launch_bounds__(512) void pool_head_kernel(const unsigned short* __restrict__ hbf,
                                                        const float* __restrict__ ln_g, const float* __restrict__ ln_b,
                                                        const float* __restrict__ hw_head, const float* __restrict__ hb_head,
                                                        float* __restrict__ out) {
    __shared__ float sred[512];
    int n = blockIdx.x, c = threadIdx.x;
    float m = -INFINITY;
    const unsigned short* p = hbf + (size_t)n * VN * CC + c;
    for (int v = 0; v < VN; ++v) m = fmaxf(m, bf2f(p[(size_t)v * CC]));
    float sum = block_reduce<512>(m, sred);
    float sq  = block_reduce<512>(m * m, sred);
    float mu = sum * (1.f / CC);
    float var = sq * (1.f / CC) - mu * mu;
    float val = (m - mu) * rsqrtf(var + LN_EPS) * ln_g[c] + ln_b[c];
    #pragma unroll
    for (int k = 0; k < NCLS; ++k) {
        float pk = block_reduce<512>(val * hw_head[c * NCLS + k], sred);
        if (c == 0) out[n * NCLS + k] = pk + hb_head[k];
    }
}

extern "C" void kernel_launch(void* const* d_in, const int* in_sizes, int n_in,
                              void* d_out, int out_size, void* d_ws, size_t ws_size,
                              hipStream_t stream) {
    const float* x       = (const float*)d_in[0];
    const float* ew      = (const float*)d_in[1];
    const float* w0      = (const float*)d_in[2];
    const float* b0      = (const float*)d_in[3];
    const float* w1      = (const float*)d_in[4];
    const float* b1      = (const float*)d_in[5];
    const float* w2      = (const float*)d_in[6];
    const float* b2      = (const float*)d_in[7];
    const float* ln_g    = (const float*)d_in[8];
    const float* ln_b    = (const float*)d_in[9];
    const float* head_w  = (const float*)d_in[10];
    const float* head_b  = (const float*)d_in[11];
    const int*   ei      = (const int*)d_in[12];
    float* out = (float*)d_out;

    size_t off = 0;
    char* base = (char*)d_ws;
    auto alloc = [&](size_t bytes) -> void* {
        void* p = base + off;
        off += (bytes + 255) & ~(size_t)255;
        return p;
    };
    int*   srcf    = (int*)  alloc((size_t)EP * 4);
    int*   tgtf    = (int*)  alloc((size_t)EP * 4);
    float* ewf     = (float*)alloc((size_t)EP * 4);
    int*   csr_src = (int*)  alloc((size_t)EP * 4);
    float* csr_w   = (float*)alloc((size_t)EP * 4);
    float* deg     = (float*)alloc((size_t)NV * 4);
    float* dinv    = (float*)alloc((size_t)NV * 4);
    int*   cnt     = (int*)  alloc((size_t)NV * 4);
    int*   cnt2    = (int*)  alloc((size_t)NV * 4);
    int*   row_ptr = (int*)  alloc((size_t)(NV + 1) * 4);
    unsigned short* wp1 = (unsigned short*)alloc((size_t)CC * CC * 2);
    unsigned short* wp2 = (unsigned short*)alloc((size_t)CC * CC * 2);
    unsigned short* hbf = (unsigned short*)alloc((size_t)NV * CC * 2);
    float* hw           = (float*)         alloc((size_t)NV * CC * 4);

    // CSR build
    zero_kernel<<<(NV + 255) / 256, 256, 0, stream>>>(deg, cnt, cnt2);
    edge_setup_kernel<<<(EP + 255) / 256, 256, 0, stream>>>(ei, ew, srcf, tgtf, ewf, deg, cnt);
    scan_kernel<<<1, 1024, 0, stream>>>(cnt, row_ptr);
    dinv_kernel<<<(NV + 255) / 256, 256, 0, stream>>>(deg, dinv);
    csr_fill_kernel<<<(EP + 255) / 256, 256, 0, stream>>>(srcf, tgtf, ewf, dinv, row_ptr, cnt2, csr_src, csr_w);

    // pack weights
    packb_kernel<<<(CC * CC + 255) / 256, 256, 0, stream>>>(w1, wp1);
    packb_kernel<<<(CC * CC + 255) / 256, 256, 0, stream>>>(w2, wp2);

    // layer 0 (rank-1 GEMM folded into scalar aggregation)
    layer0_kernel<<<NV, 256, 0, stream>>>(x, w0, b0, ln_g, ln_b, row_ptr, csr_src, csr_w, hbf);

    // layers 1..2 : WMMA GEMM + CSR aggregation + LN
    const int gemm_blocks = (NV / 64) * (CC / 32) / 8;   // 8 waves per block
    const float* biases[2] = {b1, b2};
    const unsigned short* wps[2] = {wp1, wp2};
    for (int l = 0; l < 2; ++l) {
        gemm_bf16_kernel<<<gemm_blocks, 256, 0, stream>>>(hbf, wps[l], hw);
        agg_ln_kernel<<<NV, 256, 0, stream>>>(hw, biases[l], ln_g, ln_b, row_ptr, csr_src, csr_w, hbf);
    }

    // pool + LN + head
    pool_head_kernel<<<NB, 512, 0, stream>>>(hbf, ln_g, ln_b, head_w, head_b, out);
}
